// FourierCrossAttention_40278203302085
// MI455X (gfx1250) — matrix operand order repaired and checked by reading, not compiled, not run on hardware
//
#include <hip/hip_runtime.h>
#include <hip/hip_bf16.h>

typedef __attribute__((ext_vector_type(16))) _Float16 v16h;
typedef __attribute__((ext_vector_type(8)))  _Float16 v8h;
typedef __attribute__((ext_vector_type(8)))  float    v8f;
typedef __attribute__((ext_vector_type(4)))  unsigned int u32x4;
typedef __attribute__((ext_vector_type(8)))  int      i32x8;
typedef __attribute__((ext_vector_type(4)))  int      i32x4;

#define LSEQ   2048
#define MODES  64
#define NH     8
#define NB     16
#define NE     64
#define NO     64

// ---------------------------------------------------------------------------
// TDM helper: 2D tile (elements are 2-byte halves) global -> LDS.
//   tile_w elems per row, tile_h rows, row stride in elems.
//   Descriptor bitfields per CDNA5 ISA 8.3/8.4 (count=1, type=2, data_size=1).
// ---------------------------------------------------------------------------
__device__ __forceinline__ void tdm_load_2d_f16(unsigned lds_off, const void* gptr,
                                                unsigned tile_w, unsigned tile_h,
                                                unsigned stride) {
  unsigned long long ga = (unsigned long long)gptr;
  u32x4 g0;
  g0.x = 0x1u;                                            // count=1, user mode
  g0.y = lds_off;                                         // lds_addr (bytes)
  g0.z = (unsigned)(ga & 0xFFFFFFFFu);                    // global_addr lo
  g0.w = (unsigned)((ga >> 32) & 0x01FFFFFFu) | 0x80000000u; // addr hi | type=2
  i32x8 g1;
  g1[0] = 0x00010000;                  // wg_mask=0 | data_size=1 (2B) | no pad
  g1[1] = (int)(tile_w << 16);         // ab_addr=0 | tensor_dim0[15:0]
  g1[2] = (int)(tile_h << 16);         // tensor_dim0[31:16]=0 | tensor_dim1[15:0]
  g1[3] = (int)(tile_w << 16);         // tensor_dim1[31:16]=0 | tile_dim0
  g1[4] = (int)tile_h;                 // tile_dim1 | tile_dim2=0
  g1[5] = (int)stride;                 // tensor_dim0_stride[31:0]
  g1[6] = 0;                           // stride hi | tensor_dim1_stride lo
  g1[7] = 0;
  i32x4 gz = {0, 0, 0, 0};
#if defined(__clang_major__) && (__clang_major__ >= 23)
  i32x8 gz8 = {0, 0, 0, 0, 0, 0, 0, 0};
  __builtin_amdgcn_tensor_load_to_lds(g0, g1, gz, gz, gz8, 0);
#else
  __builtin_amdgcn_tensor_load_to_lds(g0, g1, gz, gz, 0);
#endif
}

// ---------------------------------------------------------------------------
// Kernel A: build f16 DFT bases.
//   bFT[j][t] (j-major, 2048 per row): j<64 -> cos(2*pi*j*t/L), j>=64 -> -sin
//   bI [t][m] (t-major, 128 per row):  m<64 -> cos(2*pi*m*t/L), m>=64 -> +sin
// ---------------------------------------------------------------------------
__global__ __launch_bounds__(256) void basis_kernel(_Float16* __restrict__ bFT,
                                                    _Float16* __restrict__ bI) {
  int i = blockIdx.x * 256 + threadIdx.x;      // 0 .. 2048*128-1
  int t = i >> 7;
  int m = i & 127;
  int mm = m & 63;
  const float W0 = 6.28318530717958647692f / (float)LSEQ;
  float theta = (float)((mm * t) & (LSEQ - 1)) * W0;  // exact mod, L = 2^11
  float c = cosf(theta);
  float s = sinf(theta);
  bFT[m * LSEQ + t] = (_Float16)((m < 64) ? c : -s);
  bI[t * 128 + m]   = (_Float16)((m < 64) ? c :  s);
}

// ---------------------------------------------------------------------------
// Kernel B: truncated forward DFT via WMMA.
//   For one (b,h) and one of {q,k}: R[j][e] = sum_t bFT[j][t] * x[t][e]
//   A tile = basis^T [16 j x 32 t], B tile = x [32 t x 16 e], 64 K-steps.
//   Output F[(b,h)][128][64] f32 (rows 0-63 = Re, 64-127 = Im).
// ---------------------------------------------------------------------------
__global__ __launch_bounds__(256) void dft_kernel(const float* __restrict__ q,
                                                  const float* __restrict__ k,
                                                  const _Float16* __restrict__ bFT,
                                                  float* __restrict__ Fq,
                                                  float* __restrict__ Fk) {
  __shared__ _Float16 sB[128 * 32];   // basis chunk [j][t_loc]   (8 KB)
  __shared__ _Float16 sX[64 * 32];    // x^T chunk   [e][t_loc]   (4 KB)

  int idx   = blockIdx.x;
  int which = idx & 1;          // 0 = q, 1 = k
  int bh    = idx >> 1;         // b*8 + h
  int b     = bh >> 3;
  int h     = bh & 7;

  const float* src = which ? k : q;
  float* dst = (which ? Fk : Fq) + (size_t)bh * 128 * 64;
  const float* xbase = src + ((size_t)b * LSEQ * NH + h) * NE;  // + t*512 + e

  int tid  = threadIdx.x;
  int lane = tid & 31;
  int w    = tid >> 5;          // wave id 0..7 -> j-tile

  // staging maps (hoisted)
  int sj = tid >> 1, spart = tid & 1;          // basis: (j, 16 t's)
  int se = tid & 63, stg = tid >> 6;           // x: e row, 8 consecutive t's

  v8f acc[4] = {};

  for (int t0 = 0; t0 < LSEQ; t0 += 32) {
    __syncthreads();
    // stage basis chunk: one v16h copy per thread
    *(v16h*)&sB[sj * 32 + spart * 16] =
        *(const v16h*)&bFT[(size_t)sj * LSEQ + t0 + spart * 16];
    // stage x transposed: 8 coalesced dword loads -> one packed b128 LDS store
    {
      const float* col = xbase + (size_t)(t0 + stg * 8) * (NH * NE) + se;
      v8h pack;
#pragma unroll
      for (int u = 0; u < 8; ++u)
        pack[u] = (_Float16)col[(size_t)u * (NH * NE)];
      *(v8h*)&sX[se * 32 + stg * 8] = pack;
    }
    __syncthreads();

    // A fragment: row j = 16*w + (lane&15); contiguous k = 16*(lane>>4) + s, same order as B
    const _Float16* arow = &sB[((w << 4) + (lane & 15)) * 32];
    int ko = (lane >> 4) << 4;
    v8h a0 = *(const v8h*)(arow + ko);
    v8h a1 = *(const v8h*)(arow + 8 + ko);
    v16h A = __builtin_shufflevector(a0, a1, 0,1,2,3,4,5,6,7,8,9,10,11,12,13,14,15);

#pragma unroll
    for (int et = 0; et < 4; ++et) {
      const _Float16* brow = &sX[((et << 4) + (lane & 15)) * 32 + ((lane >> 4) << 4)];
      v16h B = *(const v16h*)brow;
      acc[et] = __builtin_amdgcn_wmma_f32_16x16x32_f16(
          false, A, false, B, (short)0, acc[et], false, false);
    }
  }

  // D layout: VGPR r, lanes 0-15 -> M=r, lanes 16-31 -> M=r+8; N = lane&15
#pragma unroll
  for (int et = 0; et < 4; ++et) {
#pragma unroll
    for (int r = 0; r < 8; ++r) {
      int j = (w << 4) + r + ((lane >> 4) << 3);
      int e = (et << 4) + (lane & 15);
      dst[j * 64 + e] = acc[et][r];
    }
  }
}

// ---------------------------------------------------------------------------
// Kernel C: per-(b,h) frequency-domain mixing (all in LDS, 96 KB dynamic).
//   xqk = Fq @ Fk^T (complex, K=E), complex tanh,
//   xqkv[e][x] = sum_y tanh(xqk)[x][y] * Fk[y][e],
//   coef[o][m]    =  2^{m>0} * Re( sum_e xqkv[e][m] * W[h][e][o][m] )
//   coef[o][64+m] = -2^{m>0} * Im( ... )
// ---------------------------------------------------------------------------
__global__ __launch_bounds__(256) void mix_kernel(const float* __restrict__ Fq,
                                                  const float* __restrict__ Fk,
                                                  const float* __restrict__ Wre,
                                                  const float* __restrict__ Wim,
                                                  _Float16* __restrict__ coef) {
  extern __shared__ float sm[];
  float* sFq  = sm;                 // [128][64]
  float* sFk  = sm + 128 * 64;      // [128][64]
  float* sTre = sFk + 128 * 64;     // [64][64]
  float* sTim = sTre + 64 * 64;     // [64][64]
  float* sVre = sFq;                // reuse phase-1 region
  float* sVim = sFq + 64 * 64;

  int bh  = blockIdx.x;
  int h   = bh & 7;
  int tid = threadIdx.x;

  const float* gq = Fq + (size_t)bh * 8192;
  const float* gk = Fk + (size_t)bh * 8192;
  for (int i = tid; i < 8192; i += 256) { sFq[i] = gq[i]; sFk[i] = gk[i]; }
  __syncthreads();

  // ---- xqk + complex tanh : thread handles (x = tid>>2, 16 y's) ----
  {
    int x  = tid >> 2;
    int y0 = (tid & 3) << 4;
    float tre[16], tim[16];
#pragma unroll
    for (int yy = 0; yy < 16; ++yy) { tre[yy] = 0.f; tim[yy] = 0.f; }
    for (int e = 0; e < 64; ++e) {
      float qr = sFq[x * 64 + e], qi = sFq[(64 + x) * 64 + e];
#pragma unroll
      for (int yy = 0; yy < 16; ++yy) {
        int y = y0 + yy;
        float kr = sFk[y * 64 + e], ki = sFk[(64 + y) * 64 + e];
        tre[yy] += qr * kr - qi * ki;
        tim[yy] += qr * ki + qi * kr;
      }
    }
#pragma unroll
    for (int yy = 0; yy < 16; ++yy) {
      // tanh(x+iy) = (t(1+s^2) + i s(1-t^2)) / (1 + t^2 s^2), t=tanh(x), s=tan(y)
      float t = tanhf(tre[yy]);
      float s = tanf(tim[yy]);
      float s2 = s * s;
      float rr, ii;
      if (s2 > 1e30f) {            // |tan| ~ inf (measure-zero) -> limit
        rr = (t != 0.f) ? 1.f / t : 1.f;
        ii = 0.f;
      } else {
        float den = 1.f + t * t * s2;
        rr = t * (1.f + s2) / den;
        ii = s * (1.f - t * t) / den;
      }
      sTre[x * 64 + y0 + yy] = rr;
      sTim[x * 64 + y0 + yy] = ii;
    }
  }
  __syncthreads();

  // ---- xqkv[e][x] = sum_y T[x][y] * Fk[y][e] : thread = (e, 16 x's) ----
  {
    int ee = tid >> 2;
    int x0 = (tid & 3) << 4;
    float vre[16], vim[16];
#pragma unroll
    for (int xx = 0; xx < 16; ++xx) { vre[xx] = 0.f; vim[xx] = 0.f; }
    for (int y = 0; y < 64; ++y) {
      float kr = sFk[y * 64 + ee], ki = sFk[(64 + y) * 64 + ee];
#pragma unroll
      for (int xx = 0; xx < 16; ++xx) {
        float ar = sTre[(x0 + xx) * 64 + y], ai = sTim[(x0 + xx) * 64 + y];
        vre[xx] += ar * kr - ai * ki;
        vim[xx] += ar * ki + ai * kr;
      }
    }
    __syncthreads();                 // sFq reads fully done before overwrite
#pragma unroll
    for (int xx = 0; xx < 16; ++xx) {
      sVre[ee * 64 + x0 + xx] = vre[xx];
      sVim[ee * 64 + x0 + xx] = vim[xx];
    }
  }
  __syncthreads();

  // ---- coef: thread = (o = tid>>2, 16 m's) ----
  {
    int o  = tid >> 2;
    int m0 = (tid & 3) << 4;
    const float* wr = Wre + (size_t)h * 64 * 64 * 64;   // [e][o][m]
    const float* wi = Wim + (size_t)h * 64 * 64 * 64;
    _Float16* cout = coef + (size_t)bh * 64 * 128 + o * 128;
#pragma unroll
    for (int xx = 0; xx < 16; ++xx) {
      int m = m0 + xx;
      float cr = 0.f, ci = 0.f;
      for (int e = 0; e < 64; ++e) {
        float vr = sVre[e * 64 + m], vi = sVim[e * 64 + m];
        float wre_ = wr[((size_t)e * 64 + o) * 64 + m];
        float wim_ = wi[((size_t)e * 64 + o) * 64 + m];
        cr += vr * wre_ - vi * wim_;
        ci += vr * wim_ + vi * wre_;
      }
      float wm = (m == 0) ? 1.f : 2.f;
      cout[m]      = (_Float16)(cr * wm);
      cout[64 + m] = (_Float16)(-ci * wm);
    }
  }
}

// ---------------------------------------------------------------------------
// Kernel D: truncated irFFT via WMMA, staged by the Tensor Data Mover.
//   out[o][t] = scale * sum_{m=0..127} coef[o][m] * bI[t][m]
//   One WG handles [64 o x 128 t] for one (b,h): 4 K-steps of 16x16x32.
// ---------------------------------------------------------------------------
__global__ __launch_bounds__(256) void idft_kernel(const _Float16* __restrict__ coef,
                                                   const _Float16* __restrict__ bI,
                                                   float* __restrict__ out) {
  __shared__ _Float16 sC[64 * 128];    // coef [o][m]     (16 KB)
  __shared__ _Float16 sBI[128 * 128];  // bI chunk [t][m] (32 KB)

  int blk = blockIdx.x;
  int bh  = blk >> 4;
  int t0  = (blk & 15) * 128;
  int tid = threadIdx.x;
  int lane = tid & 31;
  int w    = tid >> 5;
  int otile = w & 3;
  int thalf = w >> 2;

  // TDM staging: wave 0 issues two async tensor DMAs, waits TENSORcnt.
  if (tid < 32) {
    unsigned lds_c  = (unsigned)(unsigned long long)&sC[0];
    unsigned lds_bi = (unsigned)(unsigned long long)&sBI[0];
    // coef block: 8192 contiguous halves (tile 8192 x 1)
    tdm_load_2d_f16(lds_c, coef + (size_t)bh * 8192, 8192u, 1u, 8192u);
    // basis tile: 128 rows x 128 halves out of [2048][128]
    tdm_load_2d_f16(lds_bi, bI + (size_t)t0 * 128, 128u, 128u, 128u);
    __builtin_amdgcn_s_wait_tensorcnt(0);
  }
  __syncthreads();

  // A fragments for all 4 K-steps (row o = 16*otile + lane&15)
  const _Float16* arow = &sC[((otile << 4) + (lane & 15)) * 128];
  int ko = (lane >> 4) << 4;
  v16h Af[4];
#pragma unroll
  for (int ks = 0; ks < 4; ++ks) {
    v8h a0 = *(const v8h*)(arow + ks * 32 + ko);
    v8h a1 = *(const v8h*)(arow + ks * 32 + 8 + ko);
    Af[ks] = __builtin_shufflevector(a0, a1, 0,1,2,3,4,5,6,7,8,9,10,11,12,13,14,15);
  }

  v8f acc[4] = {};
#pragma unroll
  for (int tt = 0; tt < 4; ++tt) {
    int trow = thalf * 64 + (tt << 4) + (lane & 15);
    const _Float16* brow = &sBI[trow * 128];
#pragma unroll
    for (int ks = 0; ks < 4; ++ks) {
      v16h B = *(const v16h*)&brow[ks * 32 + ((lane >> 4) << 4)];
      acc[tt] = __builtin_amdgcn_wmma_f32_16x16x32_f16(
          false, Af[ks], false, B, (short)0, acc[tt], false, false);
    }
  }

  const float SCALE_OUT = 1.0f / (512.f * 512.f * 2048.f);  // 1/(IN*OUT*L)
  float* obase = out + (size_t)bh * NO * LSEQ;
#pragma unroll
  for (int tt = 0; tt < 4; ++tt) {
#pragma unroll
    for (int r = 0; r < 8; ++r) {
      int o = (otile << 4) + r + ((lane >> 4) << 3);
      int t = t0 + thalf * 64 + (tt << 4) + (lane & 15);
      obase[(size_t)o * LSEQ + t] = acc[tt][r] * SCALE_OUT;
    }
  }
}

// ---------------------------------------------------------------------------
// Launch. Workspace layout (bytes):
//   [0, 512K)        bFT  f16 [128][2048]
//   [512K, 1M)       bI   f16 [2048][128]
//   [1M, 5M)         Fq   f32 [128][128][64]
//   [5M, 9M)         Fk   f32 [128][128][64]
//   [9M, 11M)        coef f16 [128][64][128]
// ---------------------------------------------------------------------------
extern "C" void kernel_launch(void* const* d_in, const int* in_sizes, int n_in,
                              void* d_out, int out_size, void* d_ws, size_t ws_size,
                              hipStream_t stream) {
  const float* q   = (const float*)d_in[0];
  const float* k   = (const float*)d_in[1];
  // d_in[2] = v (unused by reference), d_in[3] = mask (unused)
  const float* Wre = (const float*)d_in[4];
  const float* Wim = (const float*)d_in[5];
  // d_in[6]/d_in[7]: index_q / index_kv == arange(64) (lowest modes)

  char* ws = (char*)d_ws;
  _Float16* bFT  = (_Float16*)(ws);
  _Float16* bI   = (_Float16*)(ws + (size_t)524288);
  float*    Fq   = (float*)   (ws + (size_t)1048576);
  float*    Fk   = (float*)   (ws + (size_t)1048576 + 4194304);
  _Float16* coef = (_Float16*)(ws + (size_t)1048576 + 8388608);
  float*    outp = (float*)d_out;

  basis_kernel<<<1024, 256, 0, stream>>>(bFT, bI);
  dft_kernel<<<2 * NB * NH, 256, 0, stream>>>(q, k, bFT, Fq, Fk);
  mix_kernel<<<NB * NH, 256, 98304, stream>>>(Fq, Fk, Wre, Wim, coef);
  idft_kernel<<<NB * NH * (LSEQ / 128), 256, 0, stream>>>(coef, bI, outp);
}
